// RGBBlock_66821101191297
// MI455X (gfx1250) — compile-verified, matrix operand order checked by reading, not run on hardware
//
#include <hip/hip_runtime.h>

typedef __attribute__((ext_vector_type(16))) _Float16 v16h;
typedef __attribute__((ext_vector_type(8)))  float    v8f;

#define C_CH   256
#define LAT    512
#define VOX    32768   // 32*32*32
#define NB     4

// ---------------------------------------------------------------------------
// Kernel A: style GEMM on the CDNA5 WMMA path.
// style(4x256) = istyle(4x512) @ W_style^T(512x256) + b_style
// coef[b][c]   = wk[c] * (style[b][c] + 1)
// One block, 512 threads = 16 waves; wave w owns a 16-channel N-tile.
// M padded 4 -> 16: pad rows clamp to row 0 and are zeroed by a mask
// (no EXEC divergence; b128 loads).  K = 512 in 16 steps of 32 (f16).
//
// Per-lane f16 layout (ISA 16-bit 16x32 A/B): element e of v16h holds
// K = (e<8 ? 0 : 16) + 8*(lane>>4) + (e&7)  -> two contiguous 8-float runs.
// ---------------------------------------------------------------------------
__global__ __launch_bounds__(512)
void style_coef_wmma(const float* __restrict__ istyle,
                     const float* __restrict__ Wst,
                     const float* __restrict__ bst,
                     const float* __restrict__ wk,
                     float* __restrict__ coef) {
  const int wave = threadIdx.x >> 5;
  const int lane = threadIdx.x & 31;
  const int nl   = lane & 15;        // M row for A / N col for B
  const int hl   = lane >> 4;        // K-half selector
  const int col  = wave * 16 + nl;   // output channel carried by this lane

  const int   rowA = (nl < NB) ? nl : 0;      // clamp padded rows to valid mem
  const float msk  = (nl < NB) ? 1.0f : 0.0f; // zero the padding

  const float* ap = istyle + rowA * LAT + 8 * hl;
  const float* bp = Wst    + col  * LAT + 8 * hl;

  v8f acc = {};
  for (int k0 = 0; k0 < LAT; k0 += 32) {
    const float4 a0 = *(const float4*)(ap + k0);       // K = k0+8h .. +3
    const float4 a1 = *(const float4*)(ap + k0 + 4);   // K = k0+8h+4 .. +7
    const float4 a2 = *(const float4*)(ap + k0 + 16);  // K = k0+16+8h ..
    const float4 a3 = *(const float4*)(ap + k0 + 20);
    const float4 b0 = *(const float4*)(bp + k0);
    const float4 b1 = *(const float4*)(bp + k0 + 4);
    const float4 b2 = *(const float4*)(bp + k0 + 16);
    const float4 b3 = *(const float4*)(bp + k0 + 20);

    v16h a, b;
    a[0]  = (_Float16)(a0.x * msk); a[1]  = (_Float16)(a0.y * msk);
    a[2]  = (_Float16)(a0.z * msk); a[3]  = (_Float16)(a0.w * msk);
    a[4]  = (_Float16)(a1.x * msk); a[5]  = (_Float16)(a1.y * msk);
    a[6]  = (_Float16)(a1.z * msk); a[7]  = (_Float16)(a1.w * msk);
    a[8]  = (_Float16)(a2.x * msk); a[9]  = (_Float16)(a2.y * msk);
    a[10] = (_Float16)(a2.z * msk); a[11] = (_Float16)(a2.w * msk);
    a[12] = (_Float16)(a3.x * msk); a[13] = (_Float16)(a3.y * msk);
    a[14] = (_Float16)(a3.z * msk); a[15] = (_Float16)(a3.w * msk);

    b[0]  = (_Float16)b0.x; b[1]  = (_Float16)b0.y;
    b[2]  = (_Float16)b0.z; b[3]  = (_Float16)b0.w;
    b[4]  = (_Float16)b1.x; b[5]  = (_Float16)b1.y;
    b[6]  = (_Float16)b1.z; b[7]  = (_Float16)b1.w;
    b[8]  = (_Float16)b2.x; b[9]  = (_Float16)b2.y;
    b[10] = (_Float16)b2.z; b[11] = (_Float16)b2.w;
    b[12] = (_Float16)b3.x; b[13] = (_Float16)b3.y;
    b[14] = (_Float16)b3.z; b[15] = (_Float16)b3.w;

    // (neg_a, A, neg_b, B, c_mod, C, reuse_a, reuse_b)
    acc = __builtin_amdgcn_wmma_f32_16x16x32_f16(false, a, false, b,
                                                 (short)0, acc, false, false);
  }
#pragma unroll
  for (int r = 0; r < 8; ++r) {
    const int m = r + 8 * hl;        // C/D layout: M = r + 8*half
    if (m < NB) {
      const float style = acc[r] + bst[col];
      coef[m * C_CH + col] = wk[col] * (style + 1.0f);
    }
  }
}

// ---------------------------------------------------------------------------
// Kernel B: bandwidth-bound channel contraction + prev_rgb add.
// tmp[b][v] = sum_c x[b][c][v] * coef[b][c] + prev[b][v]
// float4 (b128) coalesced loads; coef broadcast from LDS.
// grid (32, 4) x 256 threads, 4 voxels/thread.  This stage streams 134 MB of
// x once and sets the kernel's ~5.8 us roofline at 23.3 TB/s.
// ---------------------------------------------------------------------------
__global__ __launch_bounds__(256)
void contract_add(const float* __restrict__ x,
                  const float* __restrict__ prev,
                  const float* __restrict__ coef,
                  float* __restrict__ tmp) {
  __shared__ float sc[C_CH];
  const int b = blockIdx.y;
  sc[threadIdx.x] = coef[b * C_CH + threadIdx.x];
  __syncthreads();

  const int v = (blockIdx.x * 256 + threadIdx.x) * 4;
  const float* xb = x + (size_t)b * C_CH * VOX + v;
  float4 acc = make_float4(0.f, 0.f, 0.f, 0.f);
#pragma unroll 8
  for (int c = 0; c < C_CH; ++c) {
    const float4 xv = *(const float4*)(xb + (size_t)c * VOX);
    const float w = sc[c];
    acc.x = fmaf(xv.x, w, acc.x);
    acc.y = fmaf(xv.y, w, acc.y);
    acc.z = fmaf(xv.z, w, acc.z);
    acc.w = fmaf(xv.w, w, acc.w);
  }
  const float4 pr = *(const float4*)(prev + (size_t)b * VOX + v);
  acc.x += pr.x; acc.y += pr.y; acc.z += pr.z; acc.w += pr.w;
  *(float4*)(tmp + (size_t)b * VOX + v) = acc;
}

// ---------------------------------------------------------------------------
// Kernel C: fused trilinear 2x upsample + [1,2,1]^3/64 blur.
// blur(resize(t)) collapses to a separable 3-tap stencil on the 32^3 grid:
//   even output: 0.3125*t[j-1] + 0.625*t[j] + 0.0625*t[j+1]
//   odd  output: 0.0625*t[j-1] + 0.625*t[j] + 0.3125*t[j+1]
// (index clamping reproduces jax's boundary renormalization + edge pad).
// Each thread: 27 taps (L2-resident tmp) -> 2x2x2 output cell.
// ---------------------------------------------------------------------------
__device__ __forceinline__ int clamp31(int i) {
  return i < 0 ? 0 : (i > 31 ? 31 : i);
}

__global__ __launch_bounds__(256)
void upsample_blur(const float* __restrict__ tmp, float* __restrict__ out) {
  const int idx = blockIdx.x * 256 + threadIdx.x;   // 0 .. 4*32768-1
  const int b  = idx >> 15;
  const int r  = idx & (VOX - 1);
  const int jd = r >> 10;
  const int jh = (r >> 5) & 31;
  const int jw = r & 31;

  const float we0 = 0.3125f, we1 = 0.625f, we2 = 0.0625f; // even taps; odd = reversed

  const float* base = tmp + (size_t)b * VOX;
  float t[3][3][3];
#pragma unroll
  for (int a = 0; a < 3; ++a) {
    const int dd = clamp31(jd - 1 + a);
#pragma unroll
    for (int bb = 0; bb < 3; ++bb) {
      const int hh = clamp31(jh - 1 + bb);
#pragma unroll
      for (int cc = 0; cc < 3; ++cc) {
        const int ww = clamp31(jw - 1 + cc);
        t[a][bb][cc] = base[dd * 1024 + hh * 32 + ww];
      }
    }
  }

  float sw[3][3][2];
#pragma unroll
  for (int a = 0; a < 3; ++a)
#pragma unroll
    for (int bb = 0; bb < 3; ++bb) {
      sw[a][bb][0] = we0 * t[a][bb][0] + we1 * t[a][bb][1] + we2 * t[a][bb][2];
      sw[a][bb][1] = we2 * t[a][bb][0] + we1 * t[a][bb][1] + we0 * t[a][bb][2];
    }

  float sh[3][2][2];
#pragma unroll
  for (int a = 0; a < 3; ++a)
#pragma unroll
    for (int pw = 0; pw < 2; ++pw) {
      sh[a][0][pw] = we0 * sw[a][0][pw] + we1 * sw[a][1][pw] + we2 * sw[a][2][pw];
      sh[a][1][pw] = we2 * sw[a][0][pw] + we1 * sw[a][1][pw] + we0 * sw[a][2][pw];
    }

  float* ob = out + (size_t)b * 64 * 64 * 64;
#pragma unroll
  for (int pd = 0; pd < 2; ++pd) {
#pragma unroll
    for (int ph = 0; ph < 2; ++ph) {
      const float v0 = (pd == 0)
        ? we0 * sh[0][ph][0] + we1 * sh[1][ph][0] + we2 * sh[2][ph][0]
        : we2 * sh[0][ph][0] + we1 * sh[1][ph][0] + we0 * sh[2][ph][0];
      const float v1 = (pd == 0)
        ? we0 * sh[0][ph][1] + we1 * sh[1][ph][1] + we2 * sh[2][ph][1]
        : we2 * sh[0][ph][1] + we1 * sh[1][ph][1] + we0 * sh[2][ph][1];
      *(float2*)(ob + (2 * jd + pd) * 4096 + (2 * jh + ph) * 64 + 2 * jw) =
          make_float2(v0, v1);
    }
  }
}

// ---------------------------------------------------------------------------
extern "C" void kernel_launch(void* const* d_in, const int* in_sizes, int n_in,
                              void* d_out, int out_size, void* d_ws, size_t ws_size,
                              hipStream_t stream) {
  const float* x      = (const float*)d_in[0];  // (4,256,32,32,32)
  const float* prev   = (const float*)d_in[1];  // (4,1,32,32,32)
  const float* istyle = (const float*)d_in[2];  // (4,512)
  const float* Wst    = (const float*)d_in[3];  // (256,512)
  const float* bst    = (const float*)d_in[4];  // (256,)
  const float* wk     = (const float*)d_in[5];  // (1,256,1,1,1) -> 256 floats
  float* out = (float*)d_out;                   // (4,1,64,64,64)

  float* coef = (float*)d_ws;                   // 4*256 floats
  float* tmp  = (float*)d_ws + 1024;            // 4*32768 floats

  style_coef_wmma<<<1, 512, 0, stream>>>(istyle, Wst, bst, wk, coef);
  contract_add<<<dim3(32, 4), 256, 0, stream>>>(x, prev, coef, tmp);
  upsample_blur<<<dim3(512), 256, 0, stream>>>(tmp, out);
}
